// WindowAttention_77094662963554
// MI455X (gfx1250) — matrix operand order repaired and checked by reading, not compiled, not run on hardware
//
#include <hip/hip_runtime.h>
#include <hip/hip_bf16.h>

typedef _Float16 v16h __attribute__((ext_vector_type(16)));
typedef _Float16 v8h  __attribute__((ext_vector_type(8)));
typedef _Float16 v4h  __attribute__((ext_vector_type(4)));
typedef float    v8f  __attribute__((ext_vector_type(8)));

#define WMMA_F32_F16(a, b, c) \
  __builtin_amdgcn_wmma_f32_16x16x32_f16(false, (a), false, (b), (short)0, (c), false, false)

// ---- WMMA fragment loaders (CDNA5 16x16x32 f16 layouts, CDNA5 ISA §7.12.2) ----
// A (16x32, MxK): lane m = lane&15; halves j: K = (j<8?0:16) + (lane>=16?8:0) + (j&7)
__device__ __forceinline__ v16h load_frag_A(const _Float16* p, int row0, int stride,
                                            int k0, int lane) {
  const int m = row0 + (lane & 15);
  const int k = k0 + ((lane & 16) >> 1);           // +8 halves for upper half-wave
  const _Float16* q = p + m * stride + k;
  v8h lo = *(const v8h*)q;                          // K = k .. k+7
  v8h hi = *(const v8h*)(q + 16);                   // K = k+16 .. k+23
  return __builtin_shufflevector(lo, hi, 0,1,2,3,4,5,6,7,8,9,10,11,12,13,14,15);
}

// B (32x16, KxN) read from row-major W where B[k][n] = W[n0+n][k0+k]:
// lane n = lane&15; halves j: K = (j<8?0:16) + (lane>=16?8:0) + (j&7)
__device__ __forceinline__ v16h load_frag_B(const _Float16* w, int n0, int stride,
                                            int k0, int lane) {
  const int n = n0 + (lane & 15);
  const int k = k0 + ((lane & 16) >> 1);            // +8 halves for upper half-wave
  const _Float16* q = w + n * stride + k;
  v8h lo = *(const v8h*)q;
  v8h hi = *(const v8h*)(q + 16);
  return __builtin_shufflevector(lo, hi, 0,1,2,3,4,5,6,7,8,9,10,11,12,13,14,15);
}

// ---- prep: f32->f16 weights, gather relative-position bias into [h][i][j] ----
__global__ void swin_prep(const float* __restrict__ qkv_w, const float* __restrict__ proj_w,
                          const float* __restrict__ bias_table, const int* __restrict__ rel_index,
                          _Float16* __restrict__ qkvw_h, _Float16* __restrict__ projw_h,
                          float* __restrict__ biasg) {
  const int i = blockIdx.x * 256 + threadIdx.x;
  if (i < 49152) qkvw_h[i] = (_Float16)qkv_w[i];          // 384x128
  if (i < 16384) projw_h[i] = (_Float16)proj_w[i];        // 128x128
  if (i < 16384) {                                        // biasg[h][i][j]
    const int h = i >> 12, p = i & 4095;
    biasg[i] = bias_table[rel_index[p] * 4 + h];
  }
}

// ---- main: one block per window, 8 wave32 ----
__launch_bounds__(256, 1)
__global__ void swin_win_attn(const float* __restrict__ x, const float* __restrict__ mask,
                              const float* __restrict__ qkv_b, const float* __restrict__ proj_b,
                              const _Float16* __restrict__ qkvw_h,
                              const _Float16* __restrict__ projw_h,
                              const float* __restrict__ biasg, float* __restrict__ out) {
  extern __shared__ __align__(16) unsigned char smem[];
  _Float16* xs  = (_Float16*)(smem);             // [64][128] f16 x
  _Float16* qs  = (_Float16*)(smem + 16384);     // [64][128] f16 q*scale
  _Float16* ks  = (_Float16*)(smem + 32768);     // [64][128] f16 k
  _Float16* vts = (_Float16*)(smem + 49152);     // [128][64] f16 v transposed (chan-major)
  _Float16* os  = (_Float16*)(smem + 65536);     // [64][128] f16 attn@v result
  float*    sc  = (float*)   (smem + 81920);     // [4][64][64] f32 scores (rows reused as f16 attn)

  const int tid  = threadIdx.x;
  const int lane = tid & 31;
  const int wv   = tid >> 5;
  const int lcol = lane & 15;
  const int hr   = (lane & 16) >> 1;             // C/D fragment row offset (0 or 8)
  const size_t blk = blockIdx.x;

  // ---- stage x tile (64x128 f32 -> f16 LDS) ----
  {
    const float4* xg4 = (const float4*)(x + blk * 8192);
#pragma unroll
    for (int i = 0; i < 8; ++i) {
      float4 f = xg4[tid + i * 256];
      v4h h; h.x = (_Float16)f.x; h.y = (_Float16)f.y; h.z = (_Float16)f.z; h.w = (_Float16)f.w;
      *(v4h*)(xs + (size_t)(tid + i * 256) * 4) = h;
    }
  }
  __syncthreads();

  // ---- phase 1: QKV GEMM (64x384) = xs(64x128) . qkv_w^T ----
  // Each wave owns 3 column tiles; B fragments held in registers across the 4 row tiles
  // (4x cut in L2 weight traffic vs. re-loading per row tile).
#pragma unroll
  for (int j = 0; j < 3; ++j) {
    const int oi = wv * 3 + j;                    // 0..23
    const int o0 = oi << 4;
    v16h bfrag[4];
#pragma unroll
    for (int kk = 0; kk < 4; ++kk)
      bfrag[kk] = load_frag_B(qkvw_h, o0, 128, kk << 5, lane);
    const int oc = o0 + lcol;
    const float bb = qkv_b[oc];
#pragma unroll
    for (int mi = 0; mi < 4; ++mi) {
      v8f acc = {};
#pragma unroll
      for (int kk = 0; kk < 4; ++kk) {
        v16h a = load_frag_A(xs, mi << 4, 128, kk << 5, lane);
        acc = WMMA_F32_F16(a, bfrag[kk], acc);
      }
      if (oc < 128) {                             // q: scaled, [tok][chan]
#pragma unroll
        for (int r = 0; r < 8; ++r)
          qs[(mi * 16 + hr + r) * 128 + oc] = (_Float16)((acc[r] + bb) * 0.17677669529663687f);
      } else if (oc < 256) {                      // k: [tok][chan]
        const int c = oc - 128;
#pragma unroll
        for (int r = 0; r < 8; ++r)
          ks[(mi * 16 + hr + r) * 128 + c] = (_Float16)(acc[r] + bb);
      } else {                                    // v: transposed [chan][tok], packed 16B store
        const int c = oc - 256;
        v8h p;
#pragma unroll
        for (int r = 0; r < 8; ++r) p[r] = (_Float16)(acc[r] + bb);
        *(v8h*)(vts + c * 64 + mi * 16 + hr) = p;
      }
    }
  }
  __syncthreads();

  // ---- phase 2: scores = q . k^T per head (4 x 64x64, K=32) ----
  // wave -> (head = wv>>1, two nj column tiles); B fragment reused across 4 row tiles
  {
    const int h = wv >> 1;
    const int njb = (wv & 1) * 2;
#pragma unroll
    for (int jj = 0; jj < 2; ++jj) {
      const int nj = njb + jj;
      v16h bfrag = load_frag_B(ks, nj << 4, 128, h << 5, lane);
#pragma unroll
      for (int mi = 0; mi < 4; ++mi) {
        v16h a = load_frag_A(qs, mi << 4, 128, h << 5, lane);
        v8f acc = {};
        acc = WMMA_F32_F16(a, bfrag, acc);
        float* srow = sc + (h * 64 + mi * 16 + hr) * 64 + nj * 16 + lcol;
#pragma unroll
        for (int r = 0; r < 8; ++r) srow[r * 64] = acc[r];
      }
    }
  }
  __syncthreads();

  // ---- phase 3: softmax (one row per thread), bias + mask fused; write f16 in place ----
  {
    const int h = tid >> 6, i = tid & 63;
    float* srow = sc + (h * 64 + i) * 64;
    const float* brow = biasg + (h * 64 + i) * 64;
    const float* mrow = mask + ((size_t)(blockIdx.x & 4095) * 64 + i) * 64;
    float vals[64];
    float mx = -3.4e38f;
#pragma unroll
    for (int jx = 0; jx < 64; ++jx) { vals[jx] = srow[jx] + brow[jx] + mrow[jx]; mx = fmaxf(mx, vals[jx]); }
    float s = 0.f;
#pragma unroll
    for (int jx = 0; jx < 64; ++jx) { vals[jx] = __expf(vals[jx] - mx); s += vals[jx]; }
    const float inv = 1.0f / s;
    _Float16* arow = (_Float16*)srow;            // reuse this row's own 256B slot (stride 128 halves)
#pragma unroll
    for (int j0 = 0; j0 < 64; j0 += 8) {
      v8h p;
#pragma unroll
      for (int r = 0; r < 8; ++r) p[r] = (_Float16)(vals[j0 + r] * inv);
      *(v8h*)(arow + j0) = p;
    }
  }
  __syncthreads();

  // ---- phase 4: out = attn . v per head (64x32, K=64) ----
  // wave -> (head = wv>>1, dj = wv&1); B fragments reused across 4 row tiles
  {
    const _Float16* attnp = (const _Float16*)sc;  // f16 rows with stride 128 halves
    const int h = wv >> 1;
    const int dj = wv & 1;
    v16h bfrag[2];
#pragma unroll
    for (int kk = 0; kk < 2; ++kk)
      bfrag[kk] = load_frag_B(vts, h * 32 + dj * 16, 64, kk << 5, lane);  // vts[chan][tok]
    const int c = h * 32 + dj * 16 + lcol;
#pragma unroll
    for (int mi = 0; mi < 4; ++mi) {
      v8f acc = {};
#pragma unroll
      for (int kk = 0; kk < 2; ++kk) {
        v16h a = load_frag_A(attnp, h * 64 + mi * 16, 128, kk << 5, lane);
        acc = WMMA_F32_F16(a, bfrag[kk], acc);
      }
#pragma unroll
      for (int r = 0; r < 8; ++r)
        os[(mi * 16 + hr + r) * 128 + c] = (_Float16)acc[r];
    }
  }
  __syncthreads();

  // ---- phase 5: proj GEMM (64x128, K=128) + bias -> global f32 ----
  // wave -> one column tile; B fragments in registers across the 4 row tiles
  {
    float* og = out + blk * 8192;
    const int o0 = wv << 4;
    v16h bfrag[4];
#pragma unroll
    for (int kk = 0; kk < 4; ++kk)
      bfrag[kk] = load_frag_B(projw_h, o0, 128, kk << 5, lane);
    const int oc = o0 + lcol;
    const float pb = proj_b[oc];
#pragma unroll
    for (int mi = 0; mi < 4; ++mi) {
      v8f acc = {};
#pragma unroll
      for (int kk = 0; kk < 4; ++kk) {
        v16h a = load_frag_A(os, mi << 4, 128, kk << 5, lane);
        acc = WMMA_F32_F16(a, bfrag[kk], acc);
      }
#pragma unroll
      for (int r = 0; r < 8; ++r)
        og[(size_t)(mi * 16 + hr + r) * 128 + oc] = acc[r] + pb;
    }
  }
}

extern "C" void kernel_launch(void* const* d_in, const int* in_sizes, int n_in,
                              void* d_out, int out_size, void* d_ws, size_t ws_size,
                              hipStream_t stream) {
  (void)in_sizes; (void)n_in; (void)out_size; (void)ws_size;
  const float* x          = (const float*)d_in[0];
  const float* mask       = (const float*)d_in[1];
  const float* qkv_w      = (const float*)d_in[2];
  const float* qkv_b      = (const float*)d_in[3];
  const float* proj_w     = (const float*)d_in[4];
  const float* proj_b     = (const float*)d_in[5];
  const float* bias_table = (const float*)d_in[6];
  const int*   rel_index  = (const int*)d_in[7];
  float* out = (float*)d_out;

  _Float16* qkvw_h  = (_Float16*)d_ws;                      // 98304 B
  _Float16* projw_h = (_Float16*)((char*)d_ws + 98304);     // 32768 B
  float*    biasg   = (float*)((char*)d_ws + 131072);       // 65536 B

  swin_prep<<<192, 256, 0, stream>>>(qkv_w, proj_w, bias_table, rel_index,
                                     qkvw_h, projw_h, biasg);

  const int shmem = 147456;  // 144 KB LDS per workgroup -> 2 blocks/WGP
  hipFuncSetAttribute((const void*)swin_win_attn,
                      hipFuncAttributeMaxDynamicSharedMemorySize, shmem);
  swin_win_attn<<<16384, 256, shmem, stream>>>(x, mask, qkv_b, proj_b,
                                               qkvw_h, projw_h, biasg, out);
}
